// RPN_23845658427978
// MI455X (gfx1250) — matrix operand order repaired and checked by reading, not compiled, not run on hardware
//
#include <hip/hip_runtime.h>
#include <hip/hip_bf16.h>
#include <math.h>

typedef _Float16 v16h __attribute__((ext_vector_type(16)));
typedef _Float16 v8h  __attribute__((ext_vector_type(8)));
typedef float    v8f  __attribute__((ext_vector_type(8)));

#define NSEL   4768
#define SORTN  8192
#define MT_TOT 261888
#define MTILE  32
#define SCALE_CLAMP_F 4.135166556742356f

// -------------------------------------------------------------------------
// Kernel 1: prepack conv weights f32 -> f16 in WMMA B-fragment order.
// wB[((ntile*72 + kb)*32 + lane)*16 + j]
//   lane<16 : N = ntile*16+lane,    K = kb*32 + {0..7, 16..23}
//   lane>=16: N = ntile*16+lane-16, K = kb*32 + {8..15, 24..31}
// K ordering: k = tap*256 + c  (tap = ky*3+kx), element = conv_w[n][c][ky][kx]
// -------------------------------------------------------------------------
__global__ void rpn_prepack_kernel(const float* __restrict__ conv_w,
                                   _Float16* __restrict__ wB) {
    int o = blockIdx.x * blockDim.x + threadIdx.x;
    if (o >= 256 * 2304) return;
    int j     = o & 15;
    int L     = (o >> 4) & 31;
    int kb    = (o >> 9) % 72;
    int ntile = o / (512 * 72);
    int n   = ntile * 16 + (L & 15);
    int k   = kb * 32 + (L >> 4) * 8 + (j & 7) + (j >> 3) * 16;
    int c   = k & 255;
    int tap = k >> 8;
    wB[o] = (_Float16)conv_w[((size_t)n * 256 + c) * 9 + tap];
}

// -------------------------------------------------------------------------
// Kernel 2: fused 3x3 conv (WMMA implicit GEMM) + bias + ReLU + 1x1 heads.
// Block = 128 threads (4 waves), tile = 32 consecutive row-major cells
// (two 16x16 M-fragments per wave; each B fragment feeds 2 WMMAs, halving
// L2 weight traffic vs a 16-cell tile).
// Dynamic LDS: 32*2304 f16 patch (147456B), reused as 32x256 f32 t-buffer.
// -------------------------------------------------------------------------
__global__ __launch_bounds__(128)
void rpn_conv_head_kernel(const float* __restrict__ feat,
                          const _Float16* __restrict__ wB,
                          const float* __restrict__ conv_b,
                          const float* __restrict__ obj_w,
                          const float* __restrict__ obj_b,
                          const float* __restrict__ delta_w,
                          const float* __restrict__ delta_b,
                          float* __restrict__ scores,
                          float* __restrict__ deltas,
                          int H, int W, int lvlOff) {
    extern __shared__ unsigned char smem[];
    _Float16* Ap = (_Float16*)smem;     // [32][2304] f16 patch
    float*    tb = (float*)smem;        // [32][256]  f32 t (reused after barrier)

    const int tid   = threadIdx.x;
    const int wave  = tid >> 5;
    const int lane  = tid & 31;
    const int b     = blockIdx.y;
    const int cell0 = blockIdx.x * MTILE;
    const size_t featB = (size_t)b * 256 * H * W;

    // ---- stage A patch: 32 positions x (9 taps * 256 ch), f32 -> f16 ----
    {
        const int m  = tid & 31;   // spatial position (coalesced along w)
        const int kq = tid >> 5;   // 4 k-values per iteration
        const int cell = cell0 + m;
        const int ch = cell / W;
        const int cw = cell % W;
        for (int it = 0; it < 576; ++it) {
            int k   = it * 4 + kq;        // k = tap*256 + c
            int c   = k & 255;
            int tap = k >> 8;
            int ky  = tap / 3, kx = tap % 3;
            int gh  = ch + ky - 1;
            int gw  = cw + kx - 1;
            float v = 0.f;
            if (gh >= 0 && gh < H && gw >= 0 && gw < W)
                v = feat[featB + (size_t)c * H * W + (size_t)gh * W + gw];
            Ap[m * 2304 + k] = (_Float16)v;
        }
    }
    __syncthreads();

    // ---- implicit GEMM: M=32 spatial, N=64 per wave (4 tiles), K=2304 ----
    v8f acc[2][4] = {};
    const int m    = lane & 15;
    const int half = lane >> 4;
    for (int kb = 0; kb < 72; ++kb) {
        // A fragments (M-tiles 0,1): lane<16 holds K {k0..k0+7, k0+16..k0+23};
        // lane>=16 shifted by 8.
        union { v16h v; v8h h[2]; } a0, a1;
        const int kOff = kb * 32 + half * 8;
        {
            const int r0 = m * 2304 + kOff;
            a0.h[0] = *(const v8h*)&Ap[r0];
            a0.h[1] = *(const v8h*)&Ap[r0 + 16];
            const int r1 = (m + 16) * 2304 + kOff;
            a1.h[0] = *(const v8h*)&Ap[r1];
            a1.h[1] = *(const v8h*)&Ap[r1 + 16];
        }
#pragma unroll
        for (int t = 0; t < 4; ++t) {
            const int ntile = wave * 4 + t;
            const v16h bfrag =
                *(const v16h*)&wB[(((size_t)ntile * 72 + kb) * 32 + lane) * 16];
            acc[0][t] = __builtin_amdgcn_wmma_f32_16x16x32_f16(
                false, a0.v, false, bfrag, (short)0, acc[0][t], false, false);
            acc[1][t] = __builtin_amdgcn_wmma_f32_16x16x32_f16(
                false, a1.v, false, bfrag, (short)0, acc[1][t], false, false);
        }
    }
    __syncthreads();   // all waves done reading Ap; reuse LDS as tb

    // ---- bias + ReLU, spill t tiles to LDS ----
#pragma unroll
    for (int mt = 0; mt < 2; ++mt) {
#pragma unroll
        for (int t = 0; t < 4; ++t) {
            const int n = (wave * 4 + t) * 16 + m;
            const float bias = conv_b[n];
#pragma unroll
            for (int r = 0; r < 8; ++r) {
                const int mm = mt * 16 + r + half * 8;
                float v = acc[mt][t][r] + bias;
                tb[mm * 256 + n] = v > 0.f ? v : 0.f;
            }
        }
    }
    __syncthreads();

    // ---- 1x1 heads: 32 positions x (3 obj + 12 delta) outputs ----
    for (int o = tid; o < MTILE * 15; o += 128) {
        const int pos = o / 15;
        const int hh  = o % 15;
        const float* trow = &tb[pos * 256];
        if (hh < 3) {
            const float* wr = &obj_w[hh * 256];
            float s = obj_b[hh];
            for (int c = 0; c < 256; ++c) s += trow[c] * wr[c];
            const int gm = lvlOff + (cell0 + pos) * 3 + hh;
            scores[(size_t)b * MT_TOT + gm] = s;
        } else {
            const int d = hh - 3;
            const float* wr = &delta_w[d * 256];
            float s = delta_b[d];
            for (int c = 0; c < 256; ++c) s += trow[c] * wr[c];
            const int a2 = d >> 2, j = d & 3;
            const int gm = lvlOff + (cell0 + pos) * 3 + a2;
            deltas[((size_t)b * MT_TOT + gm) * 4 + j] = s;
        }
    }
}

// -------------------------------------------------------------------------
// Kernel 3: per (level, batch) top-k via 2048-bin radix threshold select.
// -------------------------------------------------------------------------
__global__ __launch_bounds__(256)
void rpn_topk_kernel(const float* __restrict__ scores,
                     float* __restrict__ selS, int* __restrict__ selM) {
    __shared__ int hist[2048];
    __shared__ int s_thr, s_above, s_cntHi, s_cntTie;
    const int lvl = blockIdx.x, b = blockIdx.y, tid = threadIdx.x;
    const int lvlOffA[5] = {0, 196608, 245760, 258048, 261120};
    const int lvlCntA[5] = {196608, 49152, 12288, 3072, 768};
    const int selBaseA[5] = {0, 1000, 2000, 3000, 4000};
    const int selCntA[5] = {1000, 1000, 1000, 1000, 768};
    const int N = lvlCntA[lvl], off = lvlOffA[lvl];
    const int k = selCntA[lvl], sb = selBaseA[lvl];
    const float* sc = scores + (size_t)b * MT_TOT + off;

    for (int i = tid; i < 2048; i += 256) hist[i] = 0;
    if (tid == 0) { s_cntHi = 0; s_cntTie = 0; }
    __syncthreads();
    for (int i = tid; i < N; i += 256) {
        unsigned u = __float_as_uint(sc[i]);
        unsigned key = (u & 0x80000000u) ? ~u : (u | 0x80000000u);
        atomicAdd(&hist[key >> 21], 1);
    }
    __syncthreads();
    if (tid == 0) {
        int cum = 0, t = 2047;
        for (; t > 0; --t) {
            if (cum + hist[t] >= k) break;
            cum += hist[t];
        }
        s_thr = t; s_above = cum;
    }
    __syncthreads();
    const int thr = s_thr, above = s_above;
    for (int i = tid; i < N; i += 256) {
        unsigned u = __float_as_uint(sc[i]);
        unsigned key = (u & 0x80000000u) ? ~u : (u | 0x80000000u);
        const int bin = (int)(key >> 21);
        int slot = -1;
        if (bin > thr) {
            slot = atomicAdd(&s_cntHi, 1);
        } else if (bin == thr) {
            int ts = atomicAdd(&s_cntTie, 1);
            if (ts < k - above) slot = above + ts;
        }
        if (slot >= 0 && slot < k) {
            selS[(size_t)b * NSEL + sb + slot] = sc[i];
            selM[(size_t)b * NSEL + sb + slot] = off + i;
        }
    }
}

// -------------------------------------------------------------------------
// Kernel 4: anchor generation + delta decode + clip (+ level offset boxes).
// -------------------------------------------------------------------------
__global__ void rpn_decode_kernel(const int* __restrict__ selM,
                                  const float* __restrict__ deltas,
                                  float* __restrict__ rawB,
                                  float* __restrict__ offB) {
    int g = blockIdx.x * blockDim.x + threadIdx.x;
    if (g >= 2 * NSEL) return;
    const int b = g / NSEL, slot = g % NSEL;
    int lvl = slot / 1000; if (lvl > 4) lvl = 4;
    const int lvlOffA[5] = {0, 196608, 245760, 258048, 261120};
    const int WA[5] = {256, 128, 64, 32, 16};
    const float strideA[5] = {4.f, 8.f, 16.f, 32.f, 64.f};
    const float sizeA[5] = {32.f, 64.f, 128.f, 256.f, 512.f};
    const int m = selM[g];
    const int idx = m - lvlOffA[lvl];
    const int a = idx % 3;
    const int cell = idx / 3;
    const int W = WA[lvl];
    const int wq = cell % W, hq = cell / W;
    const float ar = (a == 0) ? 0.5f : (a == 1) ? 1.f : 2.f;
    const float aw = sizeA[lvl] / sqrtf(ar);
    const float ah = aw * ar;
    const float cxa = wq * strideA[lvl];
    const float cya = hq * strideA[lvl];
    const float* d = &deltas[((size_t)b * MT_TOT + m) * 4];
    const float dx = d[0], dy = d[1];
    const float dw = fminf(d[2], SCALE_CLAMP_F);
    const float dh = fminf(d[3], SCALE_CLAMP_F);
    const float cx = dx * aw + cxa, cy = dy * ah + cya;
    const float w = expf(dw) * aw, h = expf(dh) * ah;
    const float x1 = fminf(fmaxf(cx - 0.5f * w, 0.f), 1024.f);
    const float y1 = fminf(fmaxf(cy - 0.5f * h, 0.f), 1024.f);
    const float x2 = fminf(fmaxf(cx + 0.5f * w, 0.f), 1024.f);
    const float y2 = fminf(fmaxf(cy + 0.5f * h, 0.f), 1024.f);
    const float lo = lvl * 4096.f;
    float* rb = &rawB[(size_t)g * 4];
    rb[0] = x1; rb[1] = y1; rb[2] = x2; rb[3] = y2;
    float* ob = &offB[(size_t)g * 4];
    ob[0] = x1 + lo; ob[1] = y1 + lo; ob[2] = x2 + lo; ob[3] = y2 + lo;
}

// -------------------------------------------------------------------------
// Kernel 5: per-image sort (bitonic, LDS) + greedy NMS + final top-1000.
// Dynamic LDS: keys(32K) + vals(32K) + boxes(74.5K) + area(18.6K) + sup(18.6K)
// -------------------------------------------------------------------------
__global__ __launch_bounds__(1024)
void rpn_nms_kernel(const float* __restrict__ selS,
                    const float* __restrict__ rawB,
                    const float* __restrict__ offB,
                    float* __restrict__ out) {
    extern __shared__ unsigned char sm[];
    float* keys = (float*)sm;
    int*   vals = (int*)(keys + SORTN);
    float* obx  = (float*)(vals + SORTN);
    float* area = obx + NSEL * 4;
    int*   sup  = (int*)(area + NSEL);
    const int b = blockIdx.x, tid = threadIdx.x;

    for (int i = tid; i < SORTN; i += 1024) {
        keys[i] = (i < NSEL) ? selS[(size_t)b * NSEL + i] : -3.0e38f;
        vals[i] = (i < NSEL) ? i : -1;
    }
    __syncthreads();
    // bitonic sort, descending
    for (int kk = 2; kk <= SORTN; kk <<= 1) {
        for (int jj = kk >> 1; jj > 0; jj >>= 1) {
            for (int p = tid; p < SORTN; p += 1024) {
                const int ixj = p ^ jj;
                if (ixj > p) {
                    const bool up = ((p & kk) == 0);
                    const float ka = keys[p], kb2 = keys[ixj];
                    if ((ka < kb2) == up) {
                        keys[p] = kb2; keys[ixj] = ka;
                        const int tv = vals[p]; vals[p] = vals[ixj]; vals[ixj] = tv;
                    }
                }
            }
            __syncthreads();
        }
    }
    // load sorted offset-boxes
    for (int i = tid; i < NSEL; i += 1024) {
        const int slot = vals[i];
        const float* ob = &offB[((size_t)b * NSEL + slot) * 4];
        const float x1 = ob[0], y1 = ob[1], x2 = ob[2], y2 = ob[3];
        obx[i * 4 + 0] = x1; obx[i * 4 + 1] = y1;
        obx[i * 4 + 2] = x2; obx[i * 4 + 3] = y2;
        area[i] = (x2 - x1) * (y2 - y1);
        sup[i] = 0;
    }
    __syncthreads();
    // greedy NMS: sequential over i, parallel over j
    for (int i = 0; i < NSEL; ++i) {
        if (!sup[i]) {
            const float x1 = obx[i * 4], y1 = obx[i * 4 + 1];
            const float x2 = obx[i * 4 + 2], y2 = obx[i * 4 + 3];
            const float ai = area[i];
            for (int j = i + 1 + tid; j < NSEL; j += 1024) {
                const float xx1 = fmaxf(x1, obx[j * 4]);
                const float yy1 = fmaxf(y1, obx[j * 4 + 1]);
                const float xx2 = fminf(x2, obx[j * 4 + 2]);
                const float yy2 = fminf(y2, obx[j * 4 + 3]);
                const float iw = fmaxf(xx2 - xx1, 0.f);
                const float ih = fmaxf(yy2 - yy1, 0.f);
                const float inter = iw * ih;
                const float iou = inter / (ai + area[j] - inter + 1e-9f);
                if (iou > 0.7f) sup[j] = 1;
            }
        }
        __syncthreads();
    }
    // final top-1000 of masked scores (unsuppressed first, then -1e9 fills)
    if (tid == 0) {
        int cnt = 0;
        float* op = out + (size_t)b * 1000 * 5;
        for (int j = 0; j < NSEL && cnt < 1000; ++j) {
            if (!sup[j]) {
                const float* rb = &rawB[((size_t)b * NSEL + vals[j]) * 4];
                op[cnt * 5 + 0] = rb[0]; op[cnt * 5 + 1] = rb[1];
                op[cnt * 5 + 2] = rb[2]; op[cnt * 5 + 3] = rb[3];
                op[cnt * 5 + 4] = keys[j];
                ++cnt;
            }
        }
        for (int j = 0; j < NSEL && cnt < 1000; ++j) {
            if (sup[j]) {
                const float* rb = &rawB[((size_t)b * NSEL + vals[j]) * 4];
                op[cnt * 5 + 0] = rb[0]; op[cnt * 5 + 1] = rb[1];
                op[cnt * 5 + 2] = rb[2]; op[cnt * 5 + 3] = rb[3];
                op[cnt * 5 + 4] = -1e9f;
                ++cnt;
            }
        }
    }
}

// -------------------------------------------------------------------------
extern "C" void kernel_launch(void* const* d_in, const int* in_sizes, int n_in,
                              void* d_out, int out_size, void* d_ws, size_t ws_size,
                              hipStream_t stream) {
    (void)in_sizes; (void)n_in; (void)out_size; (void)ws_size;
    const float* feats[5] = {(const float*)d_in[0], (const float*)d_in[1],
                             (const float*)d_in[2], (const float*)d_in[3],
                             (const float*)d_in[4]};
    const float* conv_w  = (const float*)d_in[5];
    const float* conv_b  = (const float*)d_in[6];
    const float* obj_w   = (const float*)d_in[7];
    const float* obj_b   = (const float*)d_in[8];
    const float* delta_w = (const float*)d_in[9];
    const float* delta_b = (const float*)d_in[10];
    float* out = (float*)d_out;

    // workspace carve-up (256B aligned)
    unsigned char* ws = (unsigned char*)d_ws;
    auto au = [](size_t x) { return (x + 255) & ~(size_t)255; };
    size_t off = 0;
    _Float16* wB    = (_Float16*)(ws + off); off = au(off + (size_t)256 * 2304 * 2);
    float* scores   = (float*)(ws + off);    off = au(off + (size_t)2 * MT_TOT * 4);
    float* deltas   = (float*)(ws + off);    off = au(off + (size_t)2 * MT_TOT * 4 * 4);
    float* selS     = (float*)(ws + off);    off = au(off + (size_t)2 * NSEL * 4);
    int*   selM     = (int*)(ws + off);      off = au(off + (size_t)2 * NSEL * 4);
    float* rawB     = (float*)(ws + off);    off = au(off + (size_t)2 * NSEL * 16);
    float* offB     = (float*)(ws + off);    off = au(off + (size_t)2 * NSEL * 16);

    // 1) weight prepack (f16 B-fragment layout)
    rpn_prepack_kernel<<<(256 * 2304 + 255) / 256, 256, 0, stream>>>(conv_w, wB);

    // 2) fused conv3x3 + ReLU + heads per level (WMMA)
    const int Hs[5] = {256, 128, 64, 32, 16};
    const int Ws[5] = {256, 128, 64, 32, 16};
    const int lvlOffA[5] = {0, 196608, 245760, 258048, 261120};
    for (int l = 0; l < 5; ++l) {
        dim3 grid(Hs[l] * Ws[l] / MTILE, 2);
        rpn_conv_head_kernel<<<grid, 128, (size_t)MTILE * 2304 * 2, stream>>>(
            feats[l], wB, conv_b, obj_w, obj_b, delta_w, delta_b,
            scores, deltas, Hs[l], Ws[l], lvlOffA[l]);
    }

    // 3) per-level top-k select
    rpn_topk_kernel<<<dim3(5, 2), 256, 0, stream>>>(scores, selS, selM);

    // 4) decode + clip + level-offset boxes
    rpn_decode_kernel<<<(2 * NSEL + 255) / 256, 256, 0, stream>>>(selM, deltas,
                                                                  rawB, offB);

    // 5) per-image sort + NMS + final top-1000
    const size_t nmsLds = (size_t)SORTN * 8 + (size_t)NSEL * (16 + 4 + 4);
    rpn_nms_kernel<<<2, 1024, nmsLds, stream>>>(selS, rawB, offB, out);
}
